// SocialLSTM_33947421507802
// MI455X (gfx1250) — matrix operand order repaired and checked by reading, not compiled, not run on hardware
//
#include <hip/hip_runtime.h>
#include <hip/hip_bf16.h>

typedef __attribute__((ext_vector_type(16))) _Float16 v16h;
typedef __attribute__((ext_vector_type(8)))  float    v8f;

#define HDIM   128
#define GCOLS  512        // 4*H gate columns
#define GRID_G 32
#define NCELLS (GRID_G*GRID_G)

__device__ __forceinline__ float sigm_(float x) { return 1.0f / (1.0f + __expf(-x)); }

__device__ __forceinline__ int cell_of(float x, float y) {
    x = fminf(fmaxf(x, -4.0f), 4.0f);
    y = fminf(fmaxf(y, -4.0f), 4.0f);
    int xi = (int)floorf((x + 4.0f) * 4.0f);   // /dx, dx = 0.25
    int yi = (int)floorf((y + 4.0f) * 4.0f);
    xi = xi < 0 ? 0 : (xi > GRID_G - 1 ? GRID_G - 1 : xi);
    yi = yi < 0 ? 0 : (yi > GRID_G - 1 ? GRID_G - 1 : yi);
    return xi * GRID_G + yi;
}

// ---------------------------------------------------------------- zero ws
__global__ __launch_bounds__(256) void social_zero_kernel(float4* p, int n4) {
    int i = blockIdx.x * 256 + threadIdx.x;
    if (i < n4) p[i] = make_float4(0.f, 0.f, 0.f, 0.f);
}

// ------------------------------------------------- LSTM cell + scatter-sum
// Block: 256 threads = 8 waves, 128 agents. LDS: W_hh as f16 (128KB) +
// fused [Wih0,Wih1,Wih2, b_ih+b_hh] per gate column (8KB).
__global__ __launch_bounds__(256) void social_lstm_kernel(
    const float* __restrict__ coords,
    const float* __restrict__ hs,
    const float* __restrict__ cs,
    const float* __restrict__ Wih,
    const float* __restrict__ Whh,
    const float* __restrict__ bih,
    const float* __restrict__ bhh,
    float* __restrict__ outC,      // c_new  (N x 128)
    float* __restrict__ perCell,   // 1024 x 128 accumulator (zeroed)
    int Ntot)
{
    extern __shared__ char smem[];
    _Float16* sW   = (_Float16*)smem;                         // 512*128 f16
    float4*   sWih = (float4*)(smem + GCOLS * HDIM * 2);      // 512 float4

    const int tid = threadIdx.x;

    // Stage W_hh -> f16 LDS (row-major [col][k])
    for (int i = tid * 4; i < GCOLS * HDIM; i += 256 * 4) {
        float4 w = *(const float4*)(Whh + i);
        sW[i + 0] = (_Float16)w.x;
        sW[i + 1] = (_Float16)w.y;
        sW[i + 2] = (_Float16)w.z;
        sW[i + 3] = (_Float16)w.w;
    }
    // Stage input weights + fused bias
    for (int c = tid; c < GCOLS; c += 256) {
        float4 wb;
        wb.x = Wih[c * 3 + 0];
        wb.y = Wih[c * 3 + 1];
        wb.z = Wih[c * 3 + 2];
        wb.w = bih[c] + bhh[c];
        sWih[c] = wb;
    }
    __syncthreads();

    const int wave    = tid >> 5;
    const int lane    = tid & 31;
    const int hi      = lane >> 4;      // 0 | 1 (lane half)
    const int lo      = lane & 15;
    const int rowBase = blockIdx.x * 128 + wave * 16;

    // ---- A fragments: 16 rows x K=128 of hidden_state, f16 ----
    v16h a[4];
    {
        int r = rowBase + lo;
        if (r >= Ntot) r = Ntot - 1;
        const float* hrow = hs + (size_t)r * HDIM;
#pragma unroll
        for (int kc = 0; kc < 4; ++kc) {
            int kb = kc * 32 + hi * 16;
            float tmp[16];
            *(float4*)(tmp + 0)  = *(const float4*)(hrow + kb);
            *(float4*)(tmp + 4)  = *(const float4*)(hrow + kb + 4);
            *(float4*)(tmp + 8)  = *(const float4*)(hrow + kb + 8);
            *(float4*)(tmp + 12) = *(const float4*)(hrow + kb + 12);
            v16h t;
#pragma unroll
            for (int u = 0; u < 16; ++u) t[u] = (_Float16)tmp[u];
            a[kc] = t;
        }
    }

    // ---- per-lane agent metadata for the 8 C-rows this lane owns ----
    float cx[8], cy[8], cz[8];
    int   cid[8], mrow[8];
    bool  ok[8];
#pragma unroll
    for (int v = 0; v < 8; ++v) {
        int m = rowBase + v + hi * 8;
        ok[v] = (m < Ntot);
        int mc = ok[v] ? m : (Ntot - 1);
        mrow[v] = mc;
        cx[v] = coords[mc * 3 + 0];
        cy[v] = coords[mc * 3 + 1];
        cz[v] = coords[mc * 3 + 2];
        cid[v] = cell_of(cx[v], cy[v]);
    }

    // ---- loop over 8 groups of 16 hidden units; gates i/f/g/o together ----
    for (int jt = 0; jt < 8; ++jt) {
        const int n = jt * 16 + lo;         // hidden-unit column of this lane
        float4 wbI = sWih[n];
        float4 wbF = sWih[n + 128];
        float4 wbG = sWih[n + 256];
        float4 wbO = sWih[n + 384];

        v8f accI, accF, accG, accO;
#pragma unroll
        for (int v = 0; v < 8; ++v) {
            accI[v] = wbI.w + wbI.x * cx[v] + wbI.y * cy[v] + wbI.z * cz[v];
            accF[v] = wbF.w + wbF.x * cx[v] + wbF.y * cy[v] + wbF.z * cz[v];
            accG[v] = wbG.w + wbG.x * cx[v] + wbG.y * cy[v] + wbG.z * cz[v];
            accO[v] = wbO.w + wbO.x * cx[v] + wbO.y * cy[v] + wbO.z * cz[v];
        }

#pragma unroll
        for (int kc = 0; kc < 4; ++kc) {
            const int koff = kc * 32 + hi * 16;
            const v16h bI = *(const v16h*)(sW + (size_t)(n)*HDIM + koff);
            const v16h bF = *(const v16h*)(sW + (size_t)(n + 128) * HDIM + koff);
            const v16h bG = *(const v16h*)(sW + (size_t)(n + 256) * HDIM + koff);
            const v16h bO = *(const v16h*)(sW + (size_t)(n + 384) * HDIM + koff);
            accI = __builtin_amdgcn_wmma_f32_16x16x32_f16(false, a[kc], false, bI, (short)0, accI, false, false);
            accF = __builtin_amdgcn_wmma_f32_16x16x32_f16(false, a[kc], false, bF, (short)0, accF, false, false);
            accG = __builtin_amdgcn_wmma_f32_16x16x32_f16(false, a[kc], false, bG, (short)0, accG, false, false);
            accO = __builtin_amdgcn_wmma_f32_16x16x32_f16(false, a[kc], false, bO, (short)0, accO, false, false);
        }

        // LSTM nonlinearity + c_new store + h_new scatter-add
#pragma unroll
        for (int v = 0; v < 8; ++v) {
            float cOld = cs[(size_t)mrow[v] * HDIM + n];
            float cNew = sigm_(accF[v]) * cOld + sigm_(accI[v]) * tanhf(accG[v]);
            float hNew = sigm_(accO[v]) * tanhf(cNew);
            if (ok[v]) {
                outC[(size_t)mrow[v] * HDIM + n] = cNew;
                unsafeAtomicAdd(perCell + (size_t)cid[v] * HDIM + n, hNew);
            }
        }
    }
}

// ------------------------------------------------- gather pooled rows
// One wave per agent: broadcast its cell's 512B row (L2-resident) to output.
__global__ __launch_bounds__(256) void social_gather_kernel(
    const float* __restrict__ coords,
    const float* __restrict__ perCell,
    float* __restrict__ outH,
    int Ntot)
{
    const int wave = threadIdx.x >> 5;
    const int lane = threadIdx.x & 31;
    const int a = blockIdx.x * 8 + wave;
    if (a >= Ntot) return;
    const int cell = cell_of(coords[a * 3 + 0], coords[a * 3 + 1]);
    float4 val = *(const float4*)(perCell + (size_t)cell * HDIM + lane * 4);
    *(float4*)(outH + (size_t)a * HDIM + lane * 4) = val;
}

extern "C" void kernel_launch(void* const* d_in, const int* in_sizes, int n_in,
                              void* d_out, int out_size, void* d_ws, size_t ws_size,
                              hipStream_t stream) {
    const float* coords = (const float*)d_in[0];
    const float* hs     = (const float*)d_in[1];
    const float* cs     = (const float*)d_in[2];
    const float* Wih    = (const float*)d_in[3];
    const float* Whh    = (const float*)d_in[4];
    const float* bih    = (const float*)d_in[5];
    const float* bhh    = (const float*)d_in[6];

    const int Ntot = in_sizes[0] / 3;            // coords is (N,3)
    float* outH = (float*)d_out;                  // h_social (N,128)
    float* outC = outH + (size_t)Ntot * HDIM;     // c_new    (N,128)
    float* perCell = (float*)d_ws;                // 1024x128 f32 = 512KB

    // 1) zero the pooling accumulator (ws is poisoned by harness)
    {
        const int n4 = NCELLS * HDIM / 4;         // 32768 float4
        social_zero_kernel<<<(n4 + 255) / 256, 256, 0, stream>>>((float4*)perCell, n4);
    }

    // 2) LSTM cell via f16 WMMA + atomic scatter-sum
    {
        const size_t smem = (size_t)GCOLS * HDIM * 2 + (size_t)GCOLS * sizeof(float4); // 128KB + 8KB
        const int nb = (Ntot + 127) / 128;
        social_lstm_kernel<<<nb, 256, smem, stream>>>(coords, hs, cs, Wih, Whh, bih, bhh,
                                                      outC, perCell, Ntot);
    }

    // 3) gather pooled hidden state
    {
        const int nb = (Ntot + 7) / 8;
        social_gather_kernel<<<nb, 256, 0, stream>>>(coords, perCell, outH, Ntot);
    }
}
